// TriangleMultiplicativeModule_33560874451051
// MI455X (gfx1250) — matrix operand order repaired and checked by reading, not compiled, not run on hardware
//
#include <hip/hip_runtime.h>
#include <hip/hip_bf16.h>

// Triangle multiplicative module ("ingoing") for MI455X / gfx1250.
// All three GEMM stages run on v_wmma_f32_16x16x32_f16 (f16 in, f32 acc).
// Pass 2 (the 512^3 x128-channel einsum) stages operand slabs into LDS with
// the Tensor Data Mover (tensor_load_to_lds + s_wait_tensorcnt), double
// buffered so the DMA for step kt+1 overlaps WMMA compute on step kt.

#define DINLINE __device__ __forceinline__

typedef _Float16 half_t;
typedef __attribute__((ext_vector_type(16))) _Float16 v16h;
typedef __attribute__((ext_vector_type(8)))  _Float16 v8h;
typedef __attribute__((ext_vector_type(8)))  float    v8f;
typedef __attribute__((ext_vector_type(4)))  float    v4f;
typedef __attribute__((ext_vector_type(4)))  unsigned int v4u;
typedef __attribute__((ext_vector_type(8)))  int      v8i;
typedef __attribute__((ext_vector_type(4)))  int      v4i;

static constexpr int    NSEQ = 512;
static constexpr int    DD   = 128;   // pair dim
static constexpr int    HH   = 128;   // hidden dim
static constexpr size_t NN   = (size_t)NSEQ * NSEQ;  // 262144 pairs

DINLINE v8f wmma_f16(v16h a, v16h b, v8f c) {
  // D = A(16x32) * B(32x16) + C, f32 accumulate
  return __builtin_amdgcn_wmma_f32_16x16x32_f16(false, a, false, b, (short)0, c,
                                                false, false);
}
DINLINE v16h cat8(v8h a, v8h b) {
  return __builtin_shufflevector(a, b, 0,1,2,3,4,5,6,7,8,9,10,11,12,13,14,15);
}
DINLINE float sigmoidf(float x) { return 1.0f / (1.0f + __expf(-x)); }

// ---------------------------------------------------------------------------
// TDM: DMA a 2-D tile (tile_dim1=128 rows x tile_dim0=32 f16) from a row-major
// [512 x 512] f16 plane (row stride 512 elements) into LDS, rows packed at
// 64B pitch.  Descriptor per CDNA5 ISA section 8.3-8.6 (D# groups 0..3).
// This toolchain exposes the 6-arg builtin:
//   (uint32x4 g0, int32x8 g1, int32x4 g2, int32x4 g3, int32x8 extra, i32 cpol)
// groups 2/3 (and the trailing group) are zero for a 2-D tensor.
// ---------------------------------------------------------------------------
DINLINE void tdm_load_tile_128x32(const half_t* gsrc, unsigned lds_byte_off) {
  const unsigned long long ga = (unsigned long long)(uintptr_t)gsrc;
  v4u g0;
  g0[0] = 1u;                                   // count=1, user descriptor
  g0[1] = lds_byte_off;                         // lds_addr[31:0]
  g0[2] = (unsigned)ga;                         // global_addr[31:0]
  g0[3] = (unsigned)((ga >> 32) & 0x01FFFFFFu)  // global_addr[56:32]
          | 0x80000000u;                        // type=2 ("image")
  v8i g1;
  g1[0] = 0x00010000;                           // wg_mask=0, data_size=1 (2B)
  g1[1] = (int)(512u << 16);                    // tensor_dim0 = 512 (lo16)
  g1[2] = (int)(512u << 16);                    // dim0 hi=0 | tensor_dim1=512
  g1[3] = (int)(32u << 16);                     // dim1 hi=0 | tile_dim0=32
  g1[4] = 128;                                  // tile_dim1=128, tile_dim2=0
  g1[5] = 512;                                  // tensor_dim0_stride = 512
  g1[6] = 0;                                    // stride hi / dim1_stride
  g1[7] = 0;
  const v4i gz4 = {0, 0, 0, 0};                 // groups 2/3 unused (2-D)
  const v8i gz8 = {0, 0, 0, 0, 0, 0, 0, 0};
  __builtin_amdgcn_tensor_load_to_lds(g0, g1, gz4, gz4, gz8, 0);
}

// ---------------------------------------------------------------------------
// Kernel 0: pack the six 128x128 weight matrices (wl, wr, wlg, wrg, wog, wo)
// into WMMA B-fragment order: per (mat, ntile, ktile) a 1KB chunk holding each
// lane's 16 halves contiguously.  B layout: lane<16 -> n=lane, K=kb+{0..7,16..23};
// lane>=16 -> n=lane-16, K=kb+{8..15,24..31}.
// ---------------------------------------------------------------------------
__global__ void pack_weights(const float* __restrict__ wl,
                             const float* __restrict__ wr,
                             const float* __restrict__ wlg,
                             const float* __restrict__ wrg,
                             const float* __restrict__ wog,
                             const float* __restrict__ wo,
                             half_t* __restrict__ packed) {
  const int lane = threadIdx.x;          // 0..31
  const int blk  = blockIdx.x;           // 0..191 = (mat*8+nt)*4+kt
  const int kt   = blk & 3;
  const int nt   = (blk >> 2) & 7;
  const int mat  = blk >> 5;
  const float* Wm[6] = {wl, wr, wlg, wrg, wog, wo};
  const float* w = Wm[mat];
  const int n     = nt * 16 + (lane & 15);
  const int kbase = kt * 32 + ((lane < 16) ? 0 : 8);
  half_t* dst = packed + ((size_t)blk * 32 + lane) * 16;
  for (int e = 0; e < 16; ++e) dst[e] = (half_t)w[(kbase + e + (e >= 8 ? 8 : 0)) * HH + n];
}

// ---------------------------------------------------------------------------
// Kernel 1: fused LayerNorm + 5 projections + masking/gating.
// Workgroup = 256 threads (8 waves) covering a 16c x 16r pair tile
// (pair p = r*512 + c; r = einsum contraction index k, c = i/j index).
// Wave w handles columns c = cbase+2w, cbase+2w+1 (two M-tiles, m = r index).
// Outputs:
//   leftKI[d][c][r], rightKI[d][c][r]  (f16, k-contiguous, via LDS transpose)
//   gateRM[p][d] = sigmoid(out-gate)   (f16, row-major)
// ---------------------------------------------------------------------------
__global__ __launch_bounds__(256) void tri_pass1(
    const float* __restrict__ x, const float* __restrict__ src_mask,
    const float* __restrict__ ln1s, const float* __restrict__ ln1b,
    const float* __restrict__ bl, const float* __restrict__ br,
    const float* __restrict__ blg, const float* __restrict__ brg,
    const float* __restrict__ bog, const half_t* __restrict__ packedW,
    half_t* __restrict__ leftKI, half_t* __restrict__ rightKI,
    half_t* __restrict__ gateRM) {
  __shared__ float s_mean[256], s_rinv[256];
  __shared__ float s_s1[DD], s_b1[DD];
  __shared__ float s_mr[16], s_mc[16];
  __shared__ alignas(32) half_t s_tr[8][2][16][16];  // per-wave transpose tiles

  const int t = threadIdx.x;
  const int wave = t >> 5, lane = t & 31;
  const int l15 = lane & 15;
  const bool lo = lane < 16;
  const int cbase = (blockIdx.x & 31) * 16;
  const int rbase = (blockIdx.x >> 5) * 16;

  if (t < DD) { s_s1[t] = ln1s[t]; s_b1[t] = ln1b[t]; }
  if (t < 16) { s_mr[t] = src_mask[rbase + t]; s_mc[t] = src_mask[cbase + t]; }

  // --- LayerNorm statistics: thread t -> pair (r = t>>4, c = t&15) ---------
  {
    const float* xr =
        x + ((size_t)(rbase + (t >> 4)) * NSEQ + (cbase + (t & 15))) * DD;
    float sum = 0.f, sq = 0.f;
#pragma unroll
    for (int i = 0; i < DD; i += 4) {
      v4f v = *(const v4f*)(xr + i);
      sum += v.x + v.y + v.z + v.w;
      sq  += v.x * v.x + v.y * v.y + v.z * v.z + v.w * v.w;
    }
    const float mean = sum * (1.f / 128.f);
    const float var  = sq * (1.f / 128.f) - mean * mean;
    s_mean[t] = mean;                    // index = r_idx*16 + c_idx
    s_rinv[t] = rsqrtf(var + 1e-6f);
  }
  __syncthreads();

  // --- Assemble A fragments (normalized x, f16) for both M-tiles ----------
  // A layout: lane<16 -> row m=l15, halves = K{kb..kb+7, kb+16..kb+23};
  //           lane>=16 -> row m=l15, halves = K{kb+8..kb+15, kb+24..kb+31}.
  v16h A[2][4];
#pragma unroll
  for (int q = 0; q < 2; ++q) {
    const int c_i = 2 * wave + q;
    const int m = l15;
    const float mean = s_mean[m * 16 + c_i];
    const float rinv = s_rinv[m * 16 + c_i];
    const float* xr = x + ((size_t)(rbase + m) * NSEQ + (cbase + c_i)) * DD;
#pragma unroll
    for (int kt = 0; kt < 4; ++kt) {
      const int k0 = kt * 32 + (lo ? 0 : 8);
      v4f x0 = *(const v4f*)(xr + k0);
      v4f x1 = *(const v4f*)(xr + k0 + 4);
      v4f x2 = *(const v4f*)(xr + k0 + 16);
      v4f x3 = *(const v4f*)(xr + k0 + 20);
      v16h a;
#pragma unroll
      for (int e = 0; e < 4; ++e) {
        a[e]      = (half_t)((x0[e] - mean) * rinv * s_s1[k0 + e]      + s_b1[k0 + e]);
        a[e + 4]  = (half_t)((x1[e] - mean) * rinv * s_s1[k0 + 4 + e]  + s_b1[k0 + 4 + e]);
        a[e + 8]  = (half_t)((x2[e] - mean) * rinv * s_s1[k0 + 16 + e] + s_b1[k0 + 16 + e]);
        a[e + 12] = (half_t)((x3[e] - mean) * rinv * s_s1[k0 + 20 + e] + s_b1[k0 + 20 + e]);
      }
      A[q][kt] = a;
    }
  }

  auto loadB = [&](int mat, int nt, int kt) -> v16h {
    return *(const v16h*)(packedW +
                          ((((size_t)mat * 8 + nt) * 4 + kt) * 32 + lane) * 16);
  };

  // --- left / right: (xn@W + b) * mask * sigmoid(xn@Wg + bg) ---------------
  for (int side = 0; side < 2; ++side) {
    const int pm = side;       // 0=wl, 1=wr
    const int gm = 2 + side;   // 2=wlg, 3=wrg
    const float* pb = side ? br : bl;
    const float* gb = side ? brg : blg;
    half_t* dst = side ? rightKI : leftKI;
    for (int nt = 0; nt < 8; ++nt) {
      v8f aP0 = {}, aP1 = {}, aG0 = {}, aG1 = {};
#pragma unroll
      for (int kt = 0; kt < 4; ++kt) {
        v16h bP = loadB(pm, nt, kt);
        v16h bG = loadB(gm, nt, kt);
        aP0 = wmma_f16(A[0][kt], bP, aP0);
        aP1 = wmma_f16(A[1][kt], bP, aP1);
        aG0 = wmma_f16(A[0][kt], bG, aG0);
        aG1 = wmma_f16(A[1][kt], bG, aG1);
      }
      const float bpn = pb[nt * 16 + l15];
      const float bgn = gb[nt * 16 + l15];
#pragma unroll
      for (int q = 0; q < 2; ++q) {
        const v8f P = q ? aP1 : aP0;
        const v8f G = q ? aG1 : aG0;
        const float mc = s_mc[2 * wave + q];
#pragma unroll
        for (int v = 0; v < 8; ++v) {
          const int m = v + (lo ? 0 : 8);
          const float val =
              (P[v] + bpn) * (s_mr[m] * mc) * sigmoidf(G[v] + bgn);
          s_tr[wave][q][l15][m] = (half_t)val;  // [d][r] mini-tile
        }
      }
      // per-wave LDS transpose read-out (same-wave LDS is in-order):
      // lane<16 handles q=0 row d=l15; lane>=16 handles q=1 row d=l15.
      {
        const int q2 = lo ? 0 : 1;
        const v16h row = *(const v16h*)&s_tr[wave][q2][l15][0];
        const size_t off = (size_t)(nt * 16 + l15) * NN +
                           (size_t)(cbase + 2 * wave + q2) * NSEQ + rbase;
        *(v16h*)(dst + off) = row;  // 16 k-contiguous halves (32B)
      }
    }
  }

  // --- out-gate: sigmoid(xn@wog + bog), row-major [p][d] -------------------
  for (int nt = 0; nt < 8; ++nt) {
    v8f aG0 = {}, aG1 = {};
#pragma unroll
    for (int kt = 0; kt < 4; ++kt) {
      v16h bG = loadB(4, nt, kt);
      aG0 = wmma_f16(A[0][kt], bG, aG0);
      aG1 = wmma_f16(A[1][kt], bG, aG1);
    }
    const float bgn = bog[nt * 16 + l15];
#pragma unroll
    for (int q = 0; q < 2; ++q) {
      const v8f G = q ? aG1 : aG0;
      const int c = cbase + 2 * wave + q;
#pragma unroll
      for (int v = 0; v < 8; ++v) {
        const int m = v + (lo ? 0 : 8);
        gateRM[((size_t)(rbase + m) * NSEQ + c) * DD + nt * 16 + l15] =
            (half_t)sigmoidf(G[v] + bgn);
      }
    }
  }
}

// ---------------------------------------------------------------------------
// Kernel 2: triangle einsum out[d][i][j] = sum_k L[d][i][k] * R[d][j][k].
// Grid: 128 channels x 4 i-blocks x 4 j-blocks (output tile 128x128, K=512).
// Operand slabs (128 rows x 32 k, 8KB f16) are DMA'd into LDS by the Tensor
// Data Mover, double buffered: wave0 stages L(kt+1), wave1 stages R(kt+1)
// while all 8 waves run WMMAs on slab kt.  TENSORcnt + barrier publishes.
// Wave grid 4x2 inside WG: each wave computes 32(i) x 64(j) = 2x4 acc tiles.
// ---------------------------------------------------------------------------
__global__ __launch_bounds__(256) void tri_pass2(
    const half_t* __restrict__ L, const half_t* __restrict__ R,
    float* __restrict__ outP) {
  __shared__ alignas(64) half_t sL[2][128 * 32];   // 2 x 8KB
  __shared__ alignas(64) half_t sR[2][128 * 32];   // 2 x 8KB

  const int t = threadIdx.x;
  const int wave = t >> 5, lane = t & 31;
  const int l15 = lane & 15;
  const bool lo = lane < 16;

  const int d  = blockIdx.x >> 4;
  const int ib = (blockIdx.x >> 2) & 3;
  const int jb = blockIdx.x & 3;
  const int i0w = (wave & 3) * 32;   // row offset within the 128-row L slab
  const int j0w = (wave >> 2) * 64;  // row offset within the 128-row R slab

  const half_t* Lblk = L + (size_t)d * NN + (size_t)(ib * 128) * NSEQ;
  const half_t* Rblk = R + (size_t)d * NN + (size_t)(jb * 128) * NSEQ;

  const bool ldrL = (wave == 0);
  const bool ldrR = (wave == 1);
  if (ldrL) tdm_load_tile_128x32(Lblk, (unsigned)(uintptr_t)&sL[0][0]);
  if (ldrR) tdm_load_tile_128x32(Rblk, (unsigned)(uintptr_t)&sR[0][0]);

  v8f acc[2][4] = {};
  for (int kt = 0; kt < 16; ++kt) {
    if (ldrL || ldrR) __builtin_amdgcn_s_wait_tensorcnt(0);
    __syncthreads();  // slab kt visible to all waves
    if (kt < 15) {    // stage kt+1 while computing kt
      const int nb = (kt + 1) & 1;
      if (ldrL)
        tdm_load_tile_128x32(Lblk + (kt + 1) * 32,
                             (unsigned)(uintptr_t)&sL[nb][0]);
      if (ldrR)
        tdm_load_tile_128x32(Rblk + (kt + 1) * 32,
                             (unsigned)(uintptr_t)&sR[nb][0]);
    }
    const half_t* bufL = &sL[kt & 1][0];
    const half_t* bufR = &sR[kt & 1][0];
    v16h a[2], b[4];
#pragma unroll
    for (int q = 0; q < 2; ++q) {
      const half_t* pa = bufL + (i0w + q * 16 + l15) * 32 + (lo ? 0 : 8);
      a[q] = cat8(*(const v8h*)pa, *(const v8h*)(pa + 16));
    }
#pragma unroll
    for (int tt = 0; tt < 4; ++tt) {
      const half_t* pb = bufR + (j0w + tt * 16 + l15) * 32 + (lo ? 0 : 8);
      b[tt] = cat8(*(const v8h*)pb, *(const v8h*)(pb + 16));
    }
#pragma unroll
    for (int q = 0; q < 2; ++q)
#pragma unroll
      for (int tt = 0; tt < 4; ++tt)
        acc[q][tt] = wmma_f16(a[q], b[tt], acc[q][tt]);
    __syncthreads();  // all reads of slab kt done before it is overwritten
  }

  float* Od = outP + (size_t)d * NN;
  const int i0 = ib * 128 + i0w;
  const int j0 = jb * 128 + j0w;
#pragma unroll
  for (int q = 0; q < 2; ++q)
#pragma unroll
    for (int tt = 0; tt < 4; ++tt) {
      const int col = j0 + tt * 16 + l15;
#pragma unroll
      for (int v = 0; v < 8; ++v) {
        const int row = i0 + q * 16 + v + (lo ? 0 : 8);
        Od[(size_t)row * NSEQ + col] = acc[q][tt][v];
      }
    }
}

// ---------------------------------------------------------------------------
// Kernel 3: LayerNorm(out) * gate, then @wo + bo via WMMA.
// WG = 256 threads <-> 256 pairs.  Stats from a first streaming pass over the
// planar einsum output; second pass normalizes+gates into an LDS-staged f16
// A fragment (per-wave private rows -> no barrier), then 16 WMMA tiles / wave.
// ---------------------------------------------------------------------------
__global__ __launch_bounds__(256) void tri_pass3(
    const float* __restrict__ outP, const half_t* __restrict__ gateRM,
    const float* __restrict__ ln2s, const float* __restrict__ ln2b,
    const half_t* __restrict__ packedW, const float* __restrict__ bo,
    float* __restrict__ y) {
  __shared__ alignas(64) half_t Ang[256][32];
  __shared__ float s_s2[DD], s_b2[DD];

  const int t = threadIdx.x;
  const int wave = t >> 5, lane = t & 31;
  const int l15 = lane & 15;
  const bool lo = lane < 16;
  const size_t p0 = (size_t)blockIdx.x * 256;
  const size_t p  = p0 + t;

  if (t < DD) { s_s2[t] = ln2s[t]; s_b2[t] = ln2b[t]; }

  float sum = 0.f, sq = 0.f;
  for (int d = 0; d < DD; ++d) {
    const float v = outP[(size_t)d * NN + p];
    sum += v; sq += v * v;
  }
  const float mean = sum * (1.f / 128.f);
  const float var  = sq * (1.f / 128.f) - mean * mean;
  const float rinv = rsqrtf(var + 1e-6f);
  __syncthreads();  // s_s2/s_b2 visible to all waves

  v8f acc[2][8] = {};
  for (int kt = 0; kt < 4; ++kt) {
    const half_t* gp = gateRM + p * DD + kt * 32;
#pragma unroll
    for (int h = 0; h < 32; ++h) {
      const int d = kt * 32 + h;
      const float v  = outP[(size_t)d * NN + p];
      const float ng = (v - mean) * rinv * s_s2[d] + s_b2[d];
      Ang[t][h] = (half_t)(ng * (float)gp[h]);
    }
    // rows [wave*32, wave*32+32) written & read by the same wave: in-order LDS
    v16h a[2];
#pragma unroll
    for (int q = 0; q < 2; ++q) {
      const half_t* pa = &Ang[wave * 32 + q * 16 + l15][lo ? 0 : 8];
      a[q] = cat8(*(const v8h*)pa, *(const v8h*)(pa + 16));
    }
#pragma unroll
    for (int nt = 0; nt < 8; ++nt) {
      const v16h b = *(const v16h*)(packedW +
          ((((size_t)5 * 8 + nt) * 4 + kt) * 32 + lane) * 16);
      acc[0][nt] = wmma_f16(a[0], b, acc[0][nt]);
      acc[1][nt] = wmma_f16(a[1], b, acc[1][nt]);
    }
  }

#pragma unroll
  for (int nt = 0; nt < 8; ++nt) {
    const float bn = bo[nt * 16 + l15];
#pragma unroll
    for (int q = 0; q < 2; ++q)
#pragma unroll
      for (int v = 0; v < 8; ++v) {
        const size_t pr = p0 + wave * 32 + q * 16 + v + (lo ? 0 : 8);
        y[pr * DD + nt * 16 + l15] = acc[q][nt][v] + bn;
      }
  }
}

// ---------------------------------------------------------------------------
extern "C" void kernel_launch(void* const* d_in, const int* in_sizes, int n_in,
                              void* d_out, int out_size, void* d_ws,
                              size_t ws_size, hipStream_t stream) {
  const float* x    = (const float*)d_in[0];
  const float* sm   = (const float*)d_in[1];
  const float* ln1s = (const float*)d_in[2];
  const float* ln1b = (const float*)d_in[3];
  const float* wl   = (const float*)d_in[4];
  const float* bl   = (const float*)d_in[5];
  const float* wr   = (const float*)d_in[6];
  const float* br   = (const float*)d_in[7];
  const float* wlg  = (const float*)d_in[8];
  const float* blg  = (const float*)d_in[9];
  const float* wrg  = (const float*)d_in[10];
  const float* brg  = (const float*)d_in[11];
  const float* wog  = (const float*)d_in[12];
  const float* bog  = (const float*)d_in[13];
  const float* ln2s = (const float*)d_in[14];
  const float* ln2b = (const float*)d_in[15];
  const float* wo   = (const float*)d_in[16];
  const float* bo   = (const float*)d_in[17];
  float* y = (float*)d_out;

  // workspace carve-up (~336 MB)
  char* ws = (char*)d_ws;
  half_t* leftKI  = (half_t*)(ws);                          // 64 MB  [d][c][r]
  half_t* rightKI = (half_t*)(ws + ((size_t)64  << 20));    // 64 MB  [d][c][r]
  half_t* gateRM  = (half_t*)(ws + ((size_t)128 << 20));    // 64 MB  [p][d]
  float*  outP    = (float*) (ws + ((size_t)192 << 20));    // 128 MB [d][i][j]
  half_t* packedW = (half_t*)(ws + ((size_t)320 << 20));    // 192 KB

  pack_weights<<<192, 32, 0, stream>>>(wl, wr, wlg, wrg, wog, wo, packedW);
  tri_pass1<<<1024, 256, 0, stream>>>(x, sm, ln1s, ln1b, bl, br, blg, brg, bog,
                                      packedW, leftKI, rightKI, gateRM);
  tri_pass2<<<2048, 256, 0, stream>>>(leftKI, rightKI, outP);
  tri_pass3<<<1024, 256, 0, stream>>>(outP, gateRM, ln2s, ln2b, packedW, bo, y);
}
